// Model_82566451298541
// MI455X (gfx1250) — matrix operand order repaired and checked by reading, not compiled, not run on hardware
//
#include <hip/hip_runtime.h>
#include <hip/hip_bf16.h>
#include <float.h>

typedef __attribute__((ext_vector_type(16))) __bf16 v16bf;
typedef __attribute__((ext_vector_type(8)))  float  v8f;
typedef __attribute__((ext_vector_type(4)))  unsigned int u32x4;
typedef __attribute__((ext_vector_type(8)))  unsigned int u32x8;

#define NB   16
#define LQ   2048
#define LK   2048
#define DD   128
#define BQ   128      // q rows per workgroup (8 waves x 16)
#define BKV  64       // kv per main-loop iteration
#define NT   (LK / BKV)

// LDS strides (bytes) — produced by TDM pad_interval/pad_amount
#define K_STRIDE 272  // 64 kv rows x 128 d bf16: 256B + 16B pad (interval=64DW, amount=4DW)
#define V_STRIDE 144  // 128 d rows x 64 kv bf16: 128B + 16B pad (interval=32DW, amount=4DW)

union FragA { v16bf v; int4 q2[2]; };

typedef __attribute__((address_space(3))) unsigned char lds_uchar;

__device__ __forceinline__ unsigned lds_addr_of(void* p) {
    return (unsigned)(unsigned long long)(lds_uchar*)p;   // flat -> LDS byte offset
}

// Issue one TDM tile copy: D# group0 built per call, group1 (dims/strides) constant.
__device__ __forceinline__ void tdm_load(unsigned lds_off, unsigned long long gva, u32x8 g1) {
    u32x4 g0;
    g0.x = 1u;                                            // count=1, user descriptor
    g0.y = lds_off;                                       // lds_addr
    g0.z = (unsigned)(gva & 0xFFFFFFFFull);               // global_addr[31:0]
    g0.w = (unsigned)((gva >> 32) & 0x1FFFFFFull)         // global_addr[56:32]
         | (2u << 30);                                    // type = 2 ("image")
    asm volatile("tensor_load_to_lds %0, %1" :: "s"(g0), "s"(g1) : "memory");
}

__device__ __forceinline__ unsigned wang_hash(unsigned s) {
    s = (s ^ 61u) ^ (s >> 16);
    s *= 9u;
    s ^= s >> 4;
    s *= 0x27d4eb2du;
    s ^= s >> 15;
    return s;
}

// ---------------- prep kernels (run once, fill d_ws) ----------------

// plain fp32 -> bf16 (K)
__global__ __launch_bounds__(256) void cvt_f32_bf16(const float* __restrict__ src,
                                                    unsigned short* __restrict__ dst) {
    const size_t i = (size_t)blockIdx.x * blockDim.x + threadIdx.x;   // one per 8 elems
    const float4* s = (const float4*)(src + i * 8);
    float4 a = s[0], b = s[1];
    float t[8] = {a.x, a.y, a.z, a.w, b.x, b.y, b.z, b.w};
    union { __bf16 h[8]; int4 q; } u;
    #pragma unroll
    for (int j = 0; j < 8; ++j) u.h[j] = (__bf16)t[j];
    *(int4*)(dst + i * 8) = u.q;
}

// Q: fold per-batch (1/inv_scale) * log2(e) into the bf16 conversion
__global__ __launch_bounds__(256) void cvt_q_scaled(const float* __restrict__ src,
                                                    unsigned short* __restrict__ dst,
                                                    const float* __restrict__ inv_scale) {
    const size_t i = (size_t)blockIdx.x * blockDim.x + threadIdx.x;   // one per 8 elems
    const int b = (int)((i * 8) / ((size_t)LQ * DD));
    const float sc = 1.44269504f / inv_scale[b];
    const float4* s = (const float4*)(src + i * 8);
    float4 a = s[0], c = s[1];
    float t[8] = {a.x, a.y, a.z, a.w, c.x, c.y, c.z, c.w};
    union { __bf16 h[8]; int4 q; } u;
    #pragma unroll
    for (int j = 0; j < 8; ++j) u.h[j] = (__bf16)(t[j] * sc);
    *(int4*)(dst + i * 8) = u.q;
}

// V [b][kv][d] fp32  ->  V^T [b][d][kv] bf16 (LDS-tiled transpose, 64-kv slabs)
__global__ __launch_bounds__(256) void vt_transpose(const float* __restrict__ v,
                                                    unsigned short* __restrict__ vt) {
    __shared__ __align__(16) unsigned char tile[64 * 272];
    const int b = blockIdx.y, kv0 = blockIdx.x * 64;
    const int t = threadIdx.x;
    {
        const int r = t >> 2, c = t & 3;                   // 64 rows x 4 d-chunks of 32
        const float* src = v + ((size_t)b * LK + kv0 + r) * DD + c * 32;
        union { __bf16 h[32]; int4 q[4]; } u;
        #pragma unroll
        for (int j = 0; j < 8; ++j) {
            float4 f = ((const float4*)src)[j];
            u.h[j*4+0] = (__bf16)f.x; u.h[j*4+1] = (__bf16)f.y;
            u.h[j*4+2] = (__bf16)f.z; u.h[j*4+3] = (__bf16)f.w;
        }
        int4* dl = (int4*)(tile + r * 272 + c * 64);
        dl[0] = u.q[0]; dl[1] = u.q[1]; dl[2] = u.q[2]; dl[3] = u.q[3];
    }
    __syncthreads();
    {
        const int d = t >> 1, h = t & 1;                   // 128 d-rows x 2 kv-halves of 32
        unsigned pk[16];
        #pragma unroll
        for (int j = 0; j < 16; ++j) {
            unsigned lo = *(const unsigned short*)(tile + (h*32 + 2*j    ) * 272 + d*2);
            unsigned hi = *(const unsigned short*)(tile + (h*32 + 2*j + 1) * 272 + d*2);
            pk[j] = lo | (hi << 16);
        }
        int4* dst = (int4*)(vt + ((size_t)b * DD + d) * LK + kv0 + h * 32);
        dst[0] = *(int4*)&pk[0];  dst[1] = *(int4*)&pk[4];
        dst[2] = *(int4*)&pk[8];  dst[3] = *(int4*)&pk[12];
    }
}

// ---------------- main flash-attention kernel ----------------

__global__ __launch_bounds__(256) void fa_fwd_kernel(
    const unsigned short* __restrict__ wsq,   // bf16 Q*(log2e/inv_scale)  [b][q][d]
    const unsigned short* __restrict__ wsk,   // bf16 K  [b][kv][d]
    const unsigned short* __restrict__ wsvt,  // bf16 V^T [b][d][kv]
    const unsigned char* __restrict__ mask,
    float* __restrict__ out)
{
    __shared__ __align__(16) unsigned char ldsK[2][BKV * K_STRIDE];
    __shared__ __align__(16) unsigned char ldsV[2][DD * V_STRIDE];

    const int tid  = threadIdx.x;
    const int lane = tid & 31;
    const int wave = tid >> 5;
    const int ln   = lane & 15;
    const int half = lane >> 4;

    const int b    = blockIdx.y;
    const int qb   = blockIdx.x * BQ + wave * 16;
    const int qcol = qb + ln;

    // ---- Q fragments (B-operand): lane = q column, K elems at d = c*32 + half*8 + {0..7, 16..23}.
    FragA qf[4];
    {
        const unsigned char* qrow = (const unsigned char*)(wsq + ((size_t)b * LQ + qcol) * DD);
        #pragma unroll
        for (int c = 0; c < 4; ++c) {
            const unsigned char* p0 = qrow + (c * 32 + half * 8) * 2;
            qf[c].q2[0] = *(const int4*)(p0);
            qf[c].q2[1] = *(const int4*)(p0 + 32);
        }
    }

    // ---- TDM descriptors (group1 constant per tensor)
    u32x8 g1k, g1v;
    // K tile: row=128 elems (2B), 64 rows; pad 16B per 256B row
    g1k[0] = (1u << 16) | (1u << 20) | (5u << 22) | (3u << 25);
    g1k[1] = (128u << 16);            // tensor_dim0 = 128
    g1k[2] = (2048u << 16);           // tensor_dim1 = 2048
    g1k[3] = (128u << 16);            // tile_dim0 = 128
    g1k[4] = 64u;                     // tile_dim1 = 64
    g1k[5] = 128u;                    // tensor_dim0_stride = 128
    g1k[6] = 0u; g1k[7] = 0u;
    // V^T tile: row=64 kv elems, 128 rows; pad 16B per 128B row
    g1v[0] = (1u << 16) | (1u << 20) | (4u << 22) | (3u << 25);
    g1v[1] = (2048u << 16);           // tensor_dim0 = 2048 (kv)
    g1v[2] = (128u << 16);            // tensor_dim1 = 128 (d)
    g1v[3] = (64u << 16);             // tile_dim0 = 64
    g1v[4] = 128u;                    // tile_dim1 = 128
    g1v[5] = 2048u;                   // tensor_dim0_stride = 2048
    g1v[6] = 0u; g1v[7] = 0u;

    const unsigned long long kbase = (unsigned long long)(wsk  + (size_t)b * LK * DD);
    const unsigned long long vbase = (unsigned long long)(wsvt + (size_t)b * DD * LK);
    unsigned koff[2] = { lds_addr_of(&ldsK[0][0]), lds_addr_of(&ldsK[1][0]) };
    unsigned voff[2] = { lds_addr_of(&ldsV[0][0]), lds_addr_of(&ldsV[1][0]) };

    // prologue: DMA tile 0 into buffer 0
    if (wave == 0) {
        tdm_load(koff[0], kbase, g1k);
        tdm_load(voff[0], vbase, g1v);
    }

    float m_prev = -1e30f;
    float l_sum  = 0.0f;
    v8f acc[8];
    #pragma unroll
    for (int dt = 0; dt < 8; ++dt) acc[dt] = (v8f){0,0,0,0,0,0,0,0};

    const unsigned char* mrow = mask + ((size_t)b * LQ + qcol) * LK + half * 8;

    // GEMM1 step order: interleave two independent subtile chains (t0/t1, then t2/t3)
    constexpr int TS1[16] = {0,1,0,1,0,1,0,1, 2,3,2,3,2,3,2,3};
    constexpr int CS1[16] = {0,0,1,1,2,2,3,3, 0,0,1,1,2,2,3,3};

    for (int kt = 0; kt < NT; ++kt) {
        const int kv0 = kt * BKV;
        const int cur = kt & 1;

        if (wave == 0) {
            if (kt + 1 < NT) {   // prefetch next tile into the other buffer
                tdm_load(koff[cur ^ 1], kbase + (size_t)(kv0 + BKV) * DD * 2, g1k);
                tdm_load(voff[cur ^ 1], vbase + (size_t)(kv0 + BKV) * 2,       g1v);
                __builtin_amdgcn_s_wait_tensorcnt(2);   // current pair landed
            } else {
                __builtin_amdgcn_s_wait_tensorcnt(0);
            }
        }
        __syncthreads();

        const unsigned char* kbuf = ldsK[cur];
        const unsigned char* vbuf = ldsV[cur];

        // mask bytes for this tile (issue early)
        uint2 mv[4];
        #pragma unroll
        for (int t = 0; t < 4; ++t) mv[t] = *(const uint2*)(mrow + kv0 + t * 16);

        // ---- GEMM1: S^T = K_tile (A) x Q (B); 16 steps, loads pipelined 2 ahead
        v8f st[4];
        FragA kfr[3];
        #pragma unroll
        for (int s = 0; s < 2; ++s) {
            const unsigned char* a = kbuf + (TS1[s] * 16 + ln) * K_STRIDE + CS1[s] * 64 + half * 16;
            kfr[s].q2[0] = *(const int4*)(a);
            kfr[s].q2[1] = *(const int4*)(a + 32);
        }
        #pragma unroll
        for (int s = 0; s < 16; ++s) {
            if (s + 2 < 16) {
                const unsigned char* a = kbuf + (TS1[s+2] * 16 + ln) * K_STRIDE + CS1[s+2] * 64 + half * 16;
                kfr[(s + 2) % 3].q2[0] = *(const int4*)(a);
                kfr[(s + 2) % 3].q2[1] = *(const int4*)(a + 32);
            }
            const int t = TS1[s], cc = CS1[s];
            v8f cin = (cc == 0) ? (v8f){0,0,0,0,0,0,0,0} : st[t];
            st[t] = __builtin_amdgcn_wmma_f32_16x16x32_bf16(
                        false, kfr[s % 3].v, false, qf[cc].v, (short)0, cin, false, false);
        }

        // ---- mask (scale already folded into Q): lane holds kv = kv0 + t*16 + half*8 + j
        #pragma unroll
        for (int t = 0; t < 4; ++t) {
            #pragma unroll
            for (int j = 0; j < 8; ++j) {
                unsigned byt = ((j < 4) ? (mv[t].x >> (8 * j)) : (mv[t].y >> (8 * (j - 4)))) & 0xFFu;
                st[t][j] = byt ? st[t][j] : -FLT_MAX;
            }
        }

        // ---- online softmax (base-2 domain): in-lane reduce + one lane-pair swap
        float mloc = -FLT_MAX;
        #pragma unroll
        for (int t = 0; t < 4; ++t)
            #pragma unroll
            for (int j = 0; j < 8; ++j) mloc = fmaxf(mloc, st[t][j]);
        float mtile = fmaxf(mloc, __shfl_xor(mloc, 16, 32));
        float mnew  = fmaxf(m_prev, mtile);
        float corr  = exp2f(m_prev - mnew);

        float dd[4][8];                    // dropout-applied P values
        float lloc = 0.0f;
        #pragma unroll
        for (int t = 0; t < 4; ++t) {
            const int kvb = kv0 + t * 16 + half * 8;
            #pragma unroll
            for (int jg = 0; jg < 2; ++jg) {           // one hash per 4 elements
                unsigned key = ((unsigned)(qcol * LK + kvb + jg * 4) >> 2)
                             + (unsigned)b * 0x9E3779B9u + 42u * 0x85EBCA6Bu;
                unsigned h = wang_hash(key);
                #pragma unroll
                for (int j2 = 0; j2 < 4; ++j2) {
                    const int j = jg * 4 + j2;
                    float p = exp2f(st[t][j] - mnew);
                    lloc += p;
                    unsigned byt = (h >> (8 * j2)) & 0xFFu;
                    dd[t][j] = (byt < 231u) ? p * (1.0f / 0.9f) : 0.0f;
                }
            }
        }
        float ltile = lloc + __shfl_xor(lloc, 16, 32);
        l_sum = l_sum * corr + ltile;

        #pragma unroll
        for (int dt = 0; dt < 8; ++dt)
            #pragma unroll
            for (int j = 0; j < 8; ++j) acc[dt][j] *= corr;

        // prefetch first two V^T fragments under the pfrag build
        FragA vfr[3];
        #pragma unroll
        for (int s = 0; s < 2; ++s) {
            const unsigned char* a = vbuf + ((s & 7) * 16 + ln) * V_STRIDE + (s >> 3) * 64 + half * 32;
            vfr[s].q2[0] = *(const int4*)(a);
            vfr[s].q2[1] = *(const int4*)(a + 16);
        }

        // ---- P^T B-fragments (one per 32-kv chunk) via one lane-pair swap each
        v16bf pfrag[2];
        #pragma unroll
        for (int c2 = 0; c2 < 2; ++c2) {
            const float* a0 = dd[c2 * 2];
            const float* a1 = dd[c2 * 2 + 1];
            #pragma unroll
            for (int j = 0; j < 8; ++j) {
                float send = half ? a0[j] : a1[j];
                float recv = __shfl_xor(send, 16, 32);
                float lo = half ? recv : a0[j];
                float hi = half ? a1[j] : recv;
                pfrag[c2][j]     = (__bf16)lo;
                pfrag[c2][8 + j] = (__bf16)hi;
            }
        }

        // ---- GEMM2: acc += V^T (A) x P^T (B); 16 steps, loads pipelined 2 ahead
        #pragma unroll
        for (int s = 0; s < 16; ++s) {
            if (s + 2 < 16) {
                const unsigned char* a = vbuf + (((s + 2) & 7) * 16 + ln) * V_STRIDE
                                              + ((s + 2) >> 3) * 64 + half * 32;
                vfr[(s + 2) % 3].q2[0] = *(const int4*)(a);
                vfr[(s + 2) % 3].q2[1] = *(const int4*)(a + 16);
            }
            acc[s & 7] = __builtin_amdgcn_wmma_f32_16x16x32_bf16(
                             false, vfr[s % 3].v, false, pfrag[s >> 3], (short)0, acc[s & 7],
                             false, false);
        }

        m_prev = mnew;
        __syncthreads();   // reads of buf[cur] done before it is re-filled next iter
    }

    // ---- epilogue
    float linv = 1.0f / l_sum;
    float* orow = out + ((size_t)b * LQ + qcol) * DD + half * 8;
    #pragma unroll
    for (int dt = 0; dt < 8; ++dt) {
        #pragma unroll
        for (int j = 0; j < 8; j += 2) {
            float2 val = make_float2(acc[dt][j] * linv, acc[dt][j + 1] * linv);
            *(float2*)(orow + dt * 16 + j) = val;
        }
    }
}

extern "C" void kernel_launch(void* const* d_in, const int* in_sizes, int n_in,
                              void* d_out, int out_size, void* d_ws, size_t ws_size,
                              hipStream_t stream) {
    const float*         q         = (const float*)d_in[0];
    const float*         k         = (const float*)d_in[1];
    const float*         v         = (const float*)d_in[2];
    const unsigned char* mask      = (const unsigned char*)d_in[3];
    const float*         inv_scale = (const float*)d_in[4];
    float*               out       = (float*)d_out;
    (void)in_sizes; (void)n_in; (void)out_size; (void)ws_size;

    const size_t nelem = (size_t)NB * LQ * DD;          // 4.19M per tensor
    unsigned short* wsq  = (unsigned short*)d_ws;
    unsigned short* wsk  = wsq + nelem;
    unsigned short* wsvt = wsk + nelem;

    // prep: bf16 conversions (+Q scale fold) + V transpose
    cvt_q_scaled<<<dim3((unsigned)(nelem / 8 / 256)), 256, 0, stream>>>(q, wsq, inv_scale);
    cvt_f32_bf16<<<dim3((unsigned)(nelem / 8 / 256)), 256, 0, stream>>>(k, wsk);
    vt_transpose<<<dim3(LK / 64, NB), 256, 0, stream>>>(v, wsvt);

    dim3 grid(LQ / BQ, NB);
    fa_fwd_kernel<<<grid, 256, 0, stream>>>(wsq, wsk, wsvt, mask, out);
}
